// BetterAuxiliaryReadout_25194278158860
// MI455X (gfx1250) — matrix operand order repaired and checked by reading, not compiled, not run on hardware
//
#include <hip/hip_runtime.h>

#define NN    500000
#define HID   256
#define AUXW  64
#define DG    576      // gate input width  (256+256+64)
#define DT    320      // trafo input width (256+64)
#define LAYER 512
#define CLS   104
#define CLSP  112      // padded to 7 x 16
#define GGR   5000
#define EPSBN 1e-5f

typedef _Float16 v4h  __attribute__((ext_vector_type(4)));
typedef _Float16 v8h  __attribute__((ext_vector_type(8)));
typedef _Float16 v16h __attribute__((ext_vector_type(16)));
typedef float    v8f  __attribute__((ext_vector_type(8)));

// ---- workspace layout (float words for header, then f16 weights) ----
#define W_SUM_IN   0        // 256
#define W_SQ_IN    256      // 256
#define W_SUM_OUT  512      // 256
#define W_SQ_OUT   768      // 256
#define W_CNT      1024     // 5120 ints
#define W_A_GATE   6144     // 576
#define W_B_GATE   6720     // 576
#define W_A_TR     7296     // 320
#define W_B_TR     7616     // 320
#define W_GATE_B   7936     // 112
#define W_T1_B     8048     // 512
#define W_T2_B     8560     // 112
#define W_HDR      8704     // header words (8704*4 = 34816 B, 32B aligned)
// f16 region offsets (in halves), base = ws + W_HDR floats
#define H_GATE_WF  0
#define H_T1_WF    (CLSP*DG)             // 64512
#define H_T2_WF    (CLSP*DG + LAYER*DT)  // 228352

#define WMMA_F16(A, B, C) \
  __builtin_amdgcn_wmma_f32_16x16x32_f16(false, (A), false, (B), (short)0, (C), false, false)

__device__ __forceinline__ v16h lds_afrag(const _Float16* p) {
  v8h lo = *(const v8h*)p;
  v8h hi = *(const v8h*)(p + 16);
  return __builtin_shufflevector(lo, hi, 0,1,2,3,4,5,6,7,8,9,10,11,12,13,14,15);
}

// ------------------------------------------------------------------
__global__ void k_zero(float* ws, float* gout) {
  int i = blockIdx.x * 256 + threadIdx.x;
  if (i < W_A_GATE) ws[i] = 0.0f;          // sums + counts (bit-zero)
  if (i < GGR * CLS) gout[i] = 0.0f;       // graph readout accumulators
}

__global__ void k_counts(const int* __restrict__ gnl, int* __restrict__ cnt) {
  int i = blockIdx.x * 256 + threadIdx.x;
  if (i < NN) atomicAdd(&cnt[gnl[i]], 1);
}

// column sums / sums-of-squares for raw_node_in (x=0) and raw_node_out (x=1)
__global__ void k_colstats(const float* __restrict__ xin,
                           const float* __restrict__ xout,
                           float* __restrict__ ws) {
  const float* X = blockIdx.x ? xout : xin;
  int col = threadIdx.x;                              // 256 columns
  long slab = (NN + (long)gridDim.y - 1) / gridDim.y;
  long r0 = (long)blockIdx.y * slab;
  long r1 = r0 + slab; if (r1 > NN) r1 = NN;
  float s = 0.0f, q = 0.0f;
  for (long r = r0; r < r1; ++r) {
    float v = X[(size_t)r * HID + col];
    s += v; q += v * v;
  }
  float* sums = ws + (blockIdx.x ? W_SUM_OUT : W_SUM_IN);
  float* sqs  = ws + (blockIdx.x ? W_SQ_OUT  : W_SQ_IN);
  unsafeAtomicAdd(&sums[col], s);
  unsafeAtomicAdd(&sqs[col],  q);
}

// per-feature BN fold coefficients: a = gamma*rsqrt(var+eps), b = beta - mean*a
__global__ void k_ab(const float* __restrict__ aux,
                     const float* __restrict__ g1, const float* __restrict__ b1,
                     const float* __restrict__ g2, const float* __restrict__ b2,
                     float* __restrict__ ws) {
  int t = threadIdx.x;
  const int* cnt = (const int*)(ws + W_CNT);
  const float invN = 1.0f / (float)NN;
  if (t < DG) {
    int k = t; float s, q;
    if (k < 256)      { s = ws[W_SUM_IN + k];        q = ws[W_SQ_IN + k]; }
    else if (k < 512) { s = ws[W_SUM_OUT + k - 256]; q = ws[W_SQ_OUT + k - 256]; }
    else {
      int j = k - 512; s = 0.0f; q = 0.0f;
      for (int g = 0; g < GGR; ++g) {
        float c = (float)cnt[g]; float v = aux[g * AUXW + j];
        s += c * v; q += c * v * v;
      }
    }
    float mean = s * invN, var = q * invN - mean * mean;
    float a = g1[k] * rsqrtf(var + EPSBN);
    ws[W_A_GATE + k] = a;
    ws[W_B_GATE + k] = b1[k] - mean * a;
  } else if (t < DG + DT) {
    int k = t - DG; float s, q;
    if (k < 256) { s = ws[W_SUM_OUT + k]; q = ws[W_SQ_OUT + k]; }
    else {
      int j = k - 256; s = 0.0f; q = 0.0f;
      for (int g = 0; g < GGR; ++g) {
        float c = (float)cnt[g]; float v = aux[g * AUXW + j];
        s += c * v; q += c * v * v;
      }
    }
    float mean = s * invN, var = q * invN - mean * mean;
    float a = g2[k] * rsqrtf(var + EPSBN);
    ws[W_A_TR + k] = a;
    ws[W_B_TR + k] = b2[k] - mean * a;
  }
}

// fold a into f16 weights, b into biases. grid (512,3), 256 thr
__global__ void k_fold(const float* __restrict__ gateW, const float* __restrict__ gateB,
                       const float* __restrict__ t1W,   const float* __restrict__ t1B,
                       const float* __restrict__ t2W,   const float* __restrict__ t2B,
                       float* __restrict__ ws) {
  _Float16* wh = (_Float16*)(ws + W_HDR);
  __shared__ float red[256];
  int row = blockIdx.x, t = threadIdx.x, which = blockIdx.y;
  if (which == 0) {
    if (row >= CLSP) return;
    _Float16* dst = wh + H_GATE_WF + (size_t)row * DG;
    float part = 0.0f;
    if (row < CLS) {
      for (int k = t; k < DG; k += 256) {
        float w = gateW[(size_t)row * DG + k];
        dst[k] = (_Float16)(w * ws[W_A_GATE + k]);
        part += w * ws[W_B_GATE + k];
      }
    } else {
      for (int k = t; k < DG; k += 256) dst[k] = (_Float16)0.0f;
    }
    red[t] = part; __syncthreads();
    for (int s = 128; s > 0; s >>= 1) { if (t < s) red[t] += red[t + s]; __syncthreads(); }
    if (t == 0) ws[W_GATE_B + row] = (row < CLS ? gateB[row] : 0.0f) + red[0];
  } else if (which == 1) {
    _Float16* dst = wh + H_T1_WF + (size_t)row * DT;
    float part = 0.0f;
    for (int k = t; k < DT; k += 256) {
      float w = t1W[(size_t)row * DT + k];
      dst[k] = (_Float16)(w * ws[W_A_TR + k]);
      part += w * ws[W_B_TR + k];
    }
    red[t] = part; __syncthreads();
    for (int s = 128; s > 0; s >>= 1) { if (t < s) red[t] += red[t + s]; __syncthreads(); }
    if (t == 0) ws[W_T1_B + row] = t1B[row] + red[0];
  } else {
    if (row >= CLSP) return;
    _Float16* dst = wh + H_T2_WF + (size_t)row * LAYER;
    for (int k = t; k < LAYER; k += 256)
      dst[k] = (row < CLS) ? (_Float16)t2W[(size_t)row * LAYER + k] : (_Float16)0.0f;
    if (t == 0) ws[W_T2_B + row] = (row < CLS) ? t2B[row] : 0.0f;
  }
}

// ------------------------------------------------------------------
// Fused main kernel: 32 nodes per block, 8 waves.
// Each wave owns ONE 16-wide output-column tile and BOTH 16-row m-tiles:
// every B fragment is loaded once and feeds two back-to-back WMMAs.
__global__ __launch_bounds__(256)
void k_main(const float* __restrict__ xin, const float* __restrict__ xout,
            const float* __restrict__ aux, const int* __restrict__ gnl,
            const float* __restrict__ ws, float* __restrict__ nodeout,
            float* __restrict__ gout) {
  __shared__ __align__(16) _Float16 Xg[32 * DG];     // f16 gate input tile
  __shared__ __align__(16) _Float16 Hc[32 * 136];    // hidden chunk (128 + pad)
  __shared__ __align__(16) _Float16 Gt[32 * CLSP];   // gating values
  __shared__ int sgid[32];

  const _Float16* wh     = (const _Float16*)(ws + W_HDR);
  const _Float16* gateWf = wh + H_GATE_WF;
  const _Float16* t1Wf   = wh + H_T1_WF;
  const _Float16* t2Wf   = wh + H_T2_WF;

  const int tid  = threadIdx.x;
  const long base = (long)blockIdx.x * 32;

  // ---- phase 1: load + convert raw rows (BN folded into weights) ----
  if (tid < 32) {
    long rowg = base + tid;
    sgid[tid] = (rowg < NN) ? gnl[rowg] : 0;
  }
  for (int i = tid; i < 2048; i += 256) {            // in/out: 32x256 each, float4
    int r = i >> 6, c4 = i & 63;
    long rowg = base + r;
    float4 vi = make_float4(0, 0, 0, 0), vo = vi;
    if (rowg < NN) {
      vi = ((const float4*)xin)[rowg * 64 + c4];
      vo = ((const float4*)xout)[rowg * 64 + c4];
    }
    v4h hi = { (_Float16)vi.x, (_Float16)vi.y, (_Float16)vi.z, (_Float16)vi.w };
    v4h ho = { (_Float16)vo.x, (_Float16)vo.y, (_Float16)vo.z, (_Float16)vo.w };
    *(v4h*)&Xg[r * DG + c4 * 4]       = hi;
    *(v4h*)&Xg[r * DG + 256 + c4 * 4] = ho;
  }
  for (int i = tid; i < 512; i += 256) {             // aux gather: 32x64
    int r = i >> 4, c4 = i & 15;
    long rowg = base + r;
    float4 va = make_float4(0, 0, 0, 0);
    if (rowg < NN) va = ((const float4*)aux)[(long)gnl[rowg] * 16 + c4];
    v4h ha = { (_Float16)va.x, (_Float16)va.y, (_Float16)va.z, (_Float16)va.w };
    *(v4h*)&Xg[r * DG + 512 + c4 * 4] = ha;
  }
  __syncthreads();

  const int wid  = tid >> 5, lane = tid & 31;
  const int half = lane >> 4, nn = lane & 15;
  const int arow0 = nn * DG;                          // A row, m-tile 0
  const int arow1 = (16 + nn) * DG;                   // A row, m-tile 1

  // ---- phase 2: gate = sigmoid([32x576] x [576x112] + b) ----
  if (wid < 7) {
    const int ct = wid;
    v8f accg0 = {0, 0, 0, 0, 0, 0, 0, 0}, accg1 = accg0;
    const _Float16* brow = gateWf + (size_t)(ct * 16 + nn) * DG;
    for (int ks = 0; ks < DG / 32; ++ks) {
      v16h b  = lds_afrag(&brow[ks * 32 + 8 * half]);
      v16h a0 = lds_afrag(&Xg[arow0 + ks * 32 + 8 * half]);
      v16h a1 = lds_afrag(&Xg[arow1 + ks * 32 + 8 * half]);
      accg0 = WMMA_F16(a0, b, accg0);
      accg1 = WMMA_F16(a1, b, accg1);
    }
    float bias = ws[W_GATE_B + ct * 16 + nn];
    for (int v = 0; v < 8; ++v) {
      float g0 = 1.0f / (1.0f + __expf(-(accg0[v] + bias)));
      float g1 = 1.0f / (1.0f + __expf(-(accg1[v] + bias)));
      Gt[(v + 8 * half) * CLSP + ct * 16 + nn]        = (_Float16)g0;
      Gt[(16 + v + 8 * half) * CLSP + ct * 16 + nn]   = (_Float16)g1;
    }
  }
  __syncthreads();

  // ---- phase 3: h = relu([32x320]x[320x512]+b) chunked; trafo += h x t2' ----
  v8f acc20 = {0, 0, 0, 0, 0, 0, 0, 0}, acc21 = acc20;
  for (int cc = 0; cc < 4; ++cc) {                   // 4 chunks of 128 hidden cols
    {                                                // all 8 waves: one ctl each, both mt
      const int ctl = wid;
      const int j = cc * 128 + ctl * 16 + nn;
      v8f acch0 = {0, 0, 0, 0, 0, 0, 0, 0}, acch1 = acch0;
      const _Float16* brow = t1Wf + (size_t)j * DT;
      for (int ks = 0; ks < DT / 32; ++ks) {
        v16h b  = lds_afrag(&brow[ks * 32 + 8 * half]);
        v16h a0 = lds_afrag(&Xg[arow0 + 256 + ks * 32 + 8 * half]);
        v16h a1 = lds_afrag(&Xg[arow1 + 256 + ks * 32 + 8 * half]);
        acch0 = WMMA_F16(a0, b, acch0);
        acch1 = WMMA_F16(a1, b, acch1);
      }
      float bias = ws[W_T1_B + j];
      for (int v = 0; v < 8; ++v) {
        float h0 = acch0[v] + bias; h0 = h0 > 0.0f ? h0 : 0.0f;
        float h1 = acch1[v] + bias; h1 = h1 > 0.0f ? h1 : 0.0f;
        Hc[(v + 8 * half) * 136 + ctl * 16 + nn]      = (_Float16)h0;
        Hc[(16 + v + 8 * half) * 136 + ctl * 16 + nn] = (_Float16)h1;
      }
    }
    __syncthreads();
    if (wid < 7) {                                   // t2 partial accumulation
      const int ct = wid;
      const _Float16* brow = t2Wf + (size_t)(ct * 16 + nn) * LAYER + cc * 128;
      for (int ks = 0; ks < 4; ++ks) {
        v16h b  = lds_afrag(&brow[ks * 32 + 8 * half]);
        v16h a0 = lds_afrag(&Hc[nn * 136 + ks * 32 + 8 * half]);
        v16h a1 = lds_afrag(&Hc[(16 + nn) * 136 + ks * 32 + 8 * half]);
        acc20 = WMMA_F16(a0, b, acc20);
        acc21 = WMMA_F16(a1, b, acc21);
      }
    }
    __syncthreads();
  }

  // ---- epilogue: nodewise = gate*trafo; segment-sum via f32 atomics ----
  if (wid < 7) {
    const int ct = wid;
    const int c = ct * 16 + nn;
    const float tb = ws[W_T2_B + c];
    for (int v = 0; v < 8; ++v) {
      for (int mt = 0; mt < 2; ++mt) {
        int m = mt * 16 + v + 8 * half;
        long rowg = base + m;
        float trafo = (mt ? acc21[v] : acc20[v]) + tb;
        float gate = (float)Gt[m * CLSP + c];
        float val = gate * trafo;
        if (rowg < NN && c < CLS) {
          nodeout[rowg * CLS + c] = val;
          unsafeAtomicAdd(&gout[(long)sgid[m] * CLS + c], val);
        }
      }
    }
  }
}

// ------------------------------------------------------------------
extern "C" void kernel_launch(void* const* d_in, const int* in_sizes, int n_in,
                              void* d_out, int out_size, void* d_ws, size_t ws_size,
                              hipStream_t stream) {
  const float* xin   = (const float*)d_in[0];
  const float* xout  = (const float*)d_in[1];
  const float* aux   = (const float*)d_in[2];
  const int*   gnl   = (const int*)d_in[3];
  // d_in[4] = num_graphs (compile-time constant GGR)
  const float* bn1g  = (const float*)d_in[5];
  const float* bn1b  = (const float*)d_in[6];
  const float* gateW = (const float*)d_in[7];
  const float* gateB = (const float*)d_in[8];
  const float* bn2g  = (const float*)d_in[9];
  const float* bn2b  = (const float*)d_in[10];
  const float* t1W   = (const float*)d_in[11];
  const float* t1b   = (const float*)d_in[12];
  const float* t2W   = (const float*)d_in[13];
  const float* t2b   = (const float*)d_in[14];

  float* out     = (float*)d_out;
  float* nodeout = out;
  float* gout    = out + (size_t)NN * CLS;
  float* ws      = (float*)d_ws;

  k_zero<<<(GGR * CLS + 255) / 256, 256, 0, stream>>>(ws, gout);
  k_counts<<<(NN + 255) / 256, 256, 0, stream>>>(gnl, (int*)(ws + W_CNT));
  k_colstats<<<dim3(2, 256), 256, 0, stream>>>(xin, xout, ws);
  k_ab<<<1, 1024, 0, stream>>>(aux, bn1g, bn1b, bn2g, bn2b, ws);
  k_fold<<<dim3(512, 3), 256, 0, stream>>>(gateW, gateB, t1W, t1b, t2W, t2b, ws);
  k_main<<<(NN + 31) / 32, 256, 0, stream>>>(xin, xout, aux, gnl, ws, nodeout, gout);
}
